// KappaModel_40295383171262
// MI455X (gfx1250) — matrix operand order repaired and checked by reading, not compiled, not run on hardware
//
#include <hip/hip_runtime.h>
#include <hip/hip_bf16.h>

// ---------------------------------------------------------------------------
// Model dims (compile-time constants, from the reference)
// ---------------------------------------------------------------------------
#define BATCH   8
#define SEQ     2048
#define MTOT    (BATCH * SEQ)   // 16384 rows
#define D_MODEL 256
#define D_INNER 1024
#define D_STATE 32
#define D_CONV  8
#define DT_RANK 16
#define XD      (DT_RANK + 2 * D_STATE)  // 80
#define HID1    64
#define IN_DIM  3
#define OUT_DIM 3
#define N_LAYERS 3

typedef __bf16 bf16;
typedef __attribute__((ext_vector_type(8)))  bf16  v8bf;
typedef __attribute__((ext_vector_type(16))) bf16  v16bf;
typedef __attribute__((ext_vector_type(8)))  float v8f;

__device__ __forceinline__ void wait_asynccnt0() {
#if __has_builtin(__builtin_amdgcn_s_wait_asynccnt)
  __builtin_amdgcn_s_wait_asynccnt(0);
#else
  asm volatile("s_wait_asynccnt 0" ::: "memory");
#endif
}

// Async DMA: 16 bytes global -> LDS per active lane (no VGPR data path).
__device__ __forceinline__ void async_copy_b128(unsigned lds_off,
                                                const void* gptr) {
  unsigned long long ga = (unsigned long long)(size_t)gptr;
  asm volatile("global_load_async_to_lds_b128 %0, %1, off"
               :: "v"(lds_off), "v"(ga)
               : "memory");
}

// ---------------------------------------------------------------------------
// Generic fp32 -> bf16 cast
// ---------------------------------------------------------------------------
__global__ void cast_bf16_kernel(const float* __restrict__ src,
                                 bf16* __restrict__ dst, int n) {
  int i = blockIdx.x * blockDim.x + threadIdx.x;
  if (i < n) dst[i] = (bf16)src[i];
}

// h -> tanh(h) as bf16 (input to each mamba layer's in_proj)
__global__ void tanh_bf16_kernel(const float* __restrict__ src,
                                 bf16* __restrict__ dst, int n) {
  int i = blockIdx.x * blockDim.x + threadIdx.x;
  if (i < n) dst[i] = (bf16)tanhf(src[i]);
}

// ---------------------------------------------------------------------------
// proj0: h[m][o] = x[m][:3] . w[o][:3] + b[o]   (block = 256 threads over o)
// ---------------------------------------------------------------------------
__global__ __launch_bounds__(256) void proj0_kernel(
    const float* __restrict__ x, const float* __restrict__ w,
    const float* __restrict__ b, float* __restrict__ h) {
  int m = blockIdx.x, o = threadIdx.x;
  const float* xr = x + (size_t)m * IN_DIM;
  float acc = b[o];
  acc = fmaf(xr[0], w[o * IN_DIM + 0], acc);
  acc = fmaf(xr[1], w[o * IN_DIM + 1], acc);
  acc = fmaf(xr[2], w[o * IN_DIM + 2], acc);
  h[(size_t)m * D_MODEL + o] = acc;
}

// ---------------------------------------------------------------------------
// bf16 NT GEMM with WMMA:  C(MxN,f32) = A(MxK,bf16) * B(NxK,bf16)^T
// block = 256 threads (8 waves), tile 128(M) x 128(N), K-step 32.
// Double-buffered LDS fed by GLOBAL_LOAD_ASYNC_TO_LDS_B128 (no VGPR staging,
// DMA of tile t+1 overlaps the WMMA compute of tile t; one barrier/K-step).
// Waves in 4(M) x 2(N) grid; each wave owns 32x64 = 2x4 wmma accumulators.
// ---------------------------------------------------------------------------
#define BM 128
#define BN 128
#define BK 32
#define LSTRIDE 48  // row stride in bf16 elems: 96B = 16B-aligned, bank-spread

__global__ __launch_bounds__(256) void gemm_bf16_nt_kernel(
    const bf16* __restrict__ A, const bf16* __restrict__ B,
    float* __restrict__ C, int M, int N, int K) {
  __shared__ bf16 sA[2][BM * LSTRIDE];
  __shared__ bf16 sB[2][BN * LSTRIDE];
  const int tid = threadIdx.x;
  const int m0 = blockIdx.y * BM;
  const int n0 = blockIdx.x * BN;
  const int wave = tid >> 5, lane = tid & 31;
  const int wm = wave & 3, wn = wave >> 2;
  const int lh = lane >> 4, lm = lane & 15;

  const int srow = tid >> 2;      // 0..63 base row (each thread: rows r, r+64)
  const int skc = (tid & 3) * 8;  // k-chunk offset (8 bf16 = 16B)

  // Issue async global->LDS DMA for one K-tile into buffer `buf`.
  auto async_stage = [&](int k0, int buf) {
#pragma unroll
    for (int i = 0; i < 2; ++i) {
      const int row = srow + i * 64;
      async_copy_b128(
          (unsigned)(size_t)&sA[buf][row * LSTRIDE + skc],
          A + (size_t)(m0 + row) * K + k0 + skc);
      if (n0 + row < N)  // unloaded rows only affect unstored columns
        async_copy_b128(
            (unsigned)(size_t)&sB[buf][row * LSTRIDE + skc],
            B + (size_t)(n0 + row) * K + k0 + skc);
    }
  };

  v8f acc[2][4] = {};

  async_stage(0, 0);
  wait_asynccnt0();
  __syncthreads();

  const int nk = K / BK;
  for (int t = 0; t < nk; ++t) {
    const int cur = t & 1;
    if (t + 1 < nk) async_stage((t + 1) * BK, 1 - cur);  // DMA next tile

    // --- fragments per documented CDNA5 VGPR layouts ---
    v16bf af[2], bfr[4];
#pragma unroll
    for (int i = 0; i < 2; ++i) {
      int r = (wm * 32 + i * 16 + lm) * LSTRIDE;
      union { v16bf v; v8bf h[2]; } u;
      u.h[0] = *(const v8bf*)&sA[cur][r + lh * 8];       // K {0..7}/{8..15}
      u.h[1] = *(const v8bf*)&sA[cur][r + 16 + lh * 8];  // K {16..23}/{24..31}
      af[i] = u.v;
    }
#pragma unroll
    for (int j = 0; j < 4; ++j) {
      int r = (wn * 64 + j * 16 + lm) * LSTRIDE;
      union { v16bf v; v8bf h[2]; } u;
      u.h[0] = *(const v8bf*)&sB[cur][r + lh * 8];       // K {0..7}/{8..15}
      u.h[1] = *(const v8bf*)&sB[cur][r + 16 + lh * 8];  // K {16..23}/{24..31}
      bfr[j] = u.v;
    }
#pragma unroll
    for (int i = 0; i < 2; ++i)
#pragma unroll
      for (int j = 0; j < 4; ++j)
        acc[i][j] = __builtin_amdgcn_wmma_f32_16x16x32_bf16(
            false, af[i], false, bfr[j], (short)0, acc[i][j], false, false);

    wait_asynccnt0();  // our DMA into the next buffer has landed
    __syncthreads();   // everyone done reading cur + next buffer published
  }

  // --- store: VGPR r -> (M = r + 8*lh, N = lm) within each 16x16 tile ---
#pragma unroll
  for (int i = 0; i < 2; ++i)
#pragma unroll
    for (int j = 0; j < 4; ++j)
#pragma unroll
      for (int r = 0; r < 8; ++r) {
        int row = m0 + wm * 32 + i * 16 + lh * 8 + r;
        int col = n0 + wn * 64 + j * 16 + lm;
        if (col < N) C[(size_t)row * N + col] = acc[i][j][r];
      }
}

// ---------------------------------------------------------------------------
// Depthwise causal conv1d (K=8) + bias + SiLU; writes f32 (scan) + bf16 (GEMM)
// xz rows are (b*L+l) of width 2*D_INNER; xin = first half.
// ---------------------------------------------------------------------------
__global__ void conv_silu_kernel(const float* __restrict__ xz,
                                 const float* __restrict__ w,
                                 const float* __restrict__ bias,
                                 float* __restrict__ xc,
                                 bf16* __restrict__ xcb, int total) {
  int idx = blockIdx.x * blockDim.x + threadIdx.x;
  if (idx >= total) return;
  int d = idx & (D_INNER - 1);
  int ml = idx >> 10;
  int l = ml & (SEQ - 1);
  float acc = bias[d];
#pragma unroll
  for (int j = 0; j < D_CONV; ++j) {
    int ls = l + j - (D_CONV - 1);
    if (ls >= 0)
      acc = fmaf(w[d * D_CONV + j],
                 xz[(size_t)(ml + j - (D_CONV - 1)) * (2 * D_INNER) + d], acc);
  }
  float s = acc / (1.f + __expf(-acc));
  xc[idx] = s;
  xcb[idx] = (bf16)s;
}

// ---------------------------------------------------------------------------
// dt = softplus(xdbl[:, :16] @ dt_w^T + dt_b)
// ---------------------------------------------------------------------------
__global__ void dtproj_kernel(const float* __restrict__ xdbl,
                              const float* __restrict__ dt_w,
                              const float* __restrict__ dt_b,
                              float* __restrict__ dt, int total) {
  int idx = blockIdx.x * blockDim.x + threadIdx.x;
  if (idx >= total) return;
  int m = idx >> 10, d = idx & (D_INNER - 1);
  float acc = dt_b[d];
#pragma unroll
  for (int r = 0; r < DT_RANK; ++r)
    acc = fmaf(xdbl[(size_t)m * XD + r], dt_w[d * DT_RANK + r], acc);
  dt[idx] = (acc > 20.f) ? acc : log1pf(__expf(acc));
}

// ---------------------------------------------------------------------------
// Selective scan: one wave32 per (b,d); lane = state index n (D_STATE==32).
// h = exp(dt*A)*h + dt*B*x ; y = (sum_n h*C + D*x) * silu(z) -> bf16
// ---------------------------------------------------------------------------
__global__ __launch_bounds__(256) void scan_kernel(
    const float* __restrict__ dt, const float* __restrict__ xdbl,
    const float* __restrict__ xc, const float* __restrict__ xz,
    const float* __restrict__ A_log, const float* __restrict__ Dskip,
    bf16* __restrict__ y) {
  const int wid = (blockIdx.x * blockDim.x + threadIdx.x) >> 5;
  const int lane = threadIdx.x & 31;
  const int b = wid >> 10, d = wid & (D_INNER - 1);
  const float Acoef = -__expf(A_log[d * D_STATE + lane]);
  const float Dd = Dskip[d];
  float h = 0.f;
  size_t m = (size_t)b * SEQ;
  for (int l = 0; l < SEQ; ++l, ++m) {
    const float dtv = dt[m * D_INNER + d];
    const float xv  = xc[m * D_INNER + d];
    const float Bv  = xdbl[m * XD + DT_RANK + lane];
    const float Cv  = xdbl[m * XD + DT_RANK + D_STATE + lane];
    const float dA  = __expf(dtv * Acoef);
    h = fmaf(dA, h, dtv * Bv * xv);
    float p = h * Cv;
#pragma unroll
    for (int off = 16; off > 0; off >>= 1)
      p += __shfl_xor(p, off, 32);
    if (lane == 0) {
      const float zv = xz[m * (2 * D_INNER) + D_INNER + d];
      const float sz = zv / (1.f + __expf(-zv));
      y[m * D_INNER + d] = (bf16)(fmaf(Dd, xv, p) * sz);
    }
  }
}

// ---------------------------------------------------------------------------
// Small dense layer (head): out[m][o] = (leaky? lrelu(in) : in) . w[o] + b[o]
// grid.x = M rows, block = OUT threads, input row staged in LDS.
// ---------------------------------------------------------------------------
__global__ void dense_kernel(const float* __restrict__ in,
                             const float* __restrict__ w,
                             const float* __restrict__ b,
                             float* __restrict__ out, int IN, int OUT,
                             int leaky) {
  __shared__ float s[D_MODEL];
  int m = blockIdx.x, o = threadIdx.x;
  for (int k = o; k < IN; k += OUT) {
    float v = in[(size_t)m * IN + k];
    if (leaky) v = v > 0.f ? v : 0.01f * v;
    s[k] = v;
  }
  __syncthreads();
  float acc = b[o];
  for (int k = 0; k < IN; ++k) acc = fmaf(s[k], w[o * IN + k], acc);
  out[(size_t)m * OUT + o] = acc;
}

// ---------------------------------------------------------------------------
// Final: out[m][:3] = proj2_w . concat(x[m], h[m], g[m]) + b
// ---------------------------------------------------------------------------
__global__ void final_kernel(const float* __restrict__ x,
                             const float* __restrict__ h,
                             const float* __restrict__ g,
                             const float* __restrict__ w,
                             const float* __restrict__ b,
                             float* __restrict__ out, int Mrows) {
  int m = blockIdx.x * blockDim.x + threadIdx.x;
  if (m >= Mrows) return;
  const int CW = IN_DIM + D_MODEL + HID1;  // 323
  float a0 = b[0], a1 = b[1], a2 = b[2];
#pragma unroll
  for (int i = 0; i < IN_DIM; ++i) {
    float v = x[(size_t)m * IN_DIM + i];
    a0 = fmaf(v, w[0 * CW + i], a0);
    a1 = fmaf(v, w[1 * CW + i], a1);
    a2 = fmaf(v, w[2 * CW + i], a2);
  }
  for (int k = 0; k < D_MODEL; ++k) {
    float v = h[(size_t)m * D_MODEL + k];
    a0 = fmaf(v, w[0 * CW + IN_DIM + k], a0);
    a1 = fmaf(v, w[1 * CW + IN_DIM + k], a1);
    a2 = fmaf(v, w[2 * CW + IN_DIM + k], a2);
  }
  for (int k = 0; k < HID1; ++k) {
    float v = g[(size_t)m * HID1 + k];
    a0 = fmaf(v, w[0 * CW + IN_DIM + D_MODEL + k], a0);
    a1 = fmaf(v, w[1 * CW + IN_DIM + D_MODEL + k], a1);
    a2 = fmaf(v, w[2 * CW + IN_DIM + D_MODEL + k], a2);
  }
  out[(size_t)m * OUT_DIM + 0] = a0;
  out[(size_t)m * OUT_DIM + 1] = a1;
  out[(size_t)m * OUT_DIM + 2] = a2;
}

// ---------------------------------------------------------------------------
// Launch
// ---------------------------------------------------------------------------
extern "C" void kernel_launch(void* const* d_in, const int* in_sizes, int n_in,
                              void* d_out, int out_size, void* d_ws,
                              size_t ws_size, hipStream_t stream) {
  (void)in_sizes; (void)n_in; (void)out_size; (void)ws_size;
  const float* x        = (const float*)d_in[0];
  const float* proj0_w  = (const float*)d_in[1];
  const float* proj0_b  = (const float*)d_in[2];
  const float* m_in_w   = (const float*)d_in[3];
  const float* m_conv_w = (const float*)d_in[4];
  const float* m_conv_b = (const float*)d_in[5];
  const float* m_xp_w   = (const float*)d_in[6];
  const float* m_dt_w   = (const float*)d_in[7];
  const float* m_dt_b   = (const float*)d_in[8];
  const float* m_A_log  = (const float*)d_in[9];
  const float* m_D      = (const float*)d_in[10];
  const float* m_out_w  = (const float*)d_in[11];
  const float* proj1_w  = (const float*)d_in[12];
  const float* proj1_b  = (const float*)d_in[13];
  const float* lin_w    = (const float*)d_in[14];
  const float* lin_b    = (const float*)d_in[15];
  const float* proj2_w  = (const float*)d_in[16];
  const float* proj2_b  = (const float*)d_in[17];
  float* out = (float*)d_out;

  // --- workspace carve-up (fixed offsets, 256B aligned) ---
  char* ws = (char*)d_ws;
  size_t off = 0;
  auto carve = [&](size_t bytes) -> void* {
    void* p = ws + off;
    off = (off + bytes + 255) & ~(size_t)255;
    return p;
  };
  float* h_f32   = (float*)carve((size_t)MTOT * D_MODEL * 4);
  bf16*  ht_b    = (bf16*) carve((size_t)MTOT * D_MODEL * 2);
  float* xz_f32  = (float*)carve((size_t)MTOT * 2 * D_INNER * 4);
  float* xc_f32  = (float*)carve((size_t)MTOT * D_INNER * 4);
  bf16*  xc_b    = (bf16*) carve((size_t)MTOT * D_INNER * 2);
  float* xdbl    = (float*)carve((size_t)MTOT * XD * 4);
  float* dt_f32  = (float*)carve((size_t)MTOT * D_INNER * 4);
  bf16*  y_b     = (bf16*) carve((size_t)MTOT * D_INNER * 2);
  float* g0      = (float*)carve((size_t)MTOT * HID1 * 4);
  float* g1      = (float*)carve((size_t)MTOT * HID1 * 4);
  bf16*  w_in_b  = (bf16*) carve((size_t)N_LAYERS * 2 * D_INNER * D_MODEL * 2);
  bf16*  w_xp_b  = (bf16*) carve((size_t)N_LAYERS * XD * D_INNER * 2);
  bf16*  w_out_b = (bf16*) carve((size_t)N_LAYERS * D_MODEL * D_INNER * 2);

  // --- cast all layer weights to bf16 once ---
  {
    int n1 = N_LAYERS * 2 * D_INNER * D_MODEL;
    int n2 = N_LAYERS * XD * D_INNER;
    int n3 = N_LAYERS * D_MODEL * D_INNER;
    cast_bf16_kernel<<<(n1 + 255) / 256, 256, 0, stream>>>(m_in_w, w_in_b, n1);
    cast_bf16_kernel<<<(n2 + 255) / 256, 256, 0, stream>>>(m_xp_w, w_xp_b, n2);
    cast_bf16_kernel<<<(n3 + 255) / 256, 256, 0, stream>>>(m_out_w, w_out_b, n3);
  }

  // --- stem ---
  proj0_kernel<<<MTOT, D_MODEL, 0, stream>>>(x, proj0_w, proj0_b, h_f32);

  const int totI = MTOT * D_INNER;
  for (int layer = 0; layer < N_LAYERS; ++layer) {
    const bf16*  Win  = w_in_b  + (size_t)layer * 2 * D_INNER * D_MODEL;
    const bf16*  Wxp  = w_xp_b  + (size_t)layer * XD * D_INNER;
    const bf16*  Wout = w_out_b + (size_t)layer * D_MODEL * D_INNER;
    const float* cw   = m_conv_w + (size_t)layer * D_INNER * D_CONV;
    const float* cb   = m_conv_b + (size_t)layer * D_INNER;
    const float* dtw  = m_dt_w  + (size_t)layer * D_INNER * DT_RANK;
    const float* dtb  = m_dt_b  + (size_t)layer * D_INNER;
    const float* Alog = m_A_log + (size_t)layer * D_INNER * D_STATE;
    const float* Dsk  = m_D     + (size_t)layer * D_INNER;

    // tanh + bf16 cast of layer input
    tanh_bf16_kernel<<<(MTOT * D_MODEL) / 256, 256, 0, stream>>>(
        h_f32, ht_b, MTOT * D_MODEL);

    // xz = ht @ in_w^T   (16384 x 2048 x 256)
    {
      dim3 g((2 * D_INNER) / BN, MTOT / BM);
      gemm_bf16_nt_kernel<<<g, 256, 0, stream>>>(ht_b, Win, xz_f32, MTOT,
                                                 2 * D_INNER, D_MODEL);
    }
    // causal depthwise conv + SiLU
    conv_silu_kernel<<<totI / 256, 256, 0, stream>>>(xz_f32, cw, cb, xc_f32,
                                                     xc_b, totI);
    // xdbl = xc @ xproj_w^T   (16384 x 80 x 1024)
    {
      dim3 g((XD + BN - 1) / BN, MTOT / BM);
      gemm_bf16_nt_kernel<<<g, 256, 0, stream>>>(xc_b, Wxp, xdbl, MTOT, XD,
                                                 D_INNER);
    }
    // dt = softplus(xdbl[:,:16] @ dt_w^T + dt_b)
    dtproj_kernel<<<totI / 256, 256, 0, stream>>>(xdbl, dtw, dtb, dt_f32, totI);
    // selective scan (8192 waves, 8 waves/block)
    scan_kernel<<<(BATCH * D_INNER) / 8, 256, 0, stream>>>(
        dt_f32, xdbl, xc_f32, xz_f32, Alog, Dsk, y_b);
    // h = y @ out_w^T   (16384 x 256 x 1024)
    {
      dim3 g(D_MODEL / BN, MTOT / BM);
      gemm_bf16_nt_kernel<<<g, 256, 0, stream>>>(y_b, Wout, h_f32, MTOT,
                                                 D_MODEL, D_INNER);
    }
  }

  // --- head ---
  dense_kernel<<<MTOT, HID1, 0, stream>>>(h_f32, proj1_w, proj1_b, g0, D_MODEL,
                                          HID1, 0);
  dense_kernel<<<MTOT, HID1, 0, stream>>>(g0, lin_w, lin_b, g1, HID1, HID1, 1);
  dense_kernel<<<MTOT, HID1, 0, stream>>>(g1, lin_w + HID1 * HID1,
                                          lin_b + HID1, g0, HID1, HID1, 1);
  final_kernel<<<(MTOT + 255) / 256, 256, 0, stream>>>(x, h_f32, g0, proj2_w,
                                                       proj2_b, out, MTOT);
}
